// shap_GOSE_model_82437602280208
// MI455X (gfx1250) — matrix operand order repaired and checked by reading, not compiled, not run on hardware
//
#include <hip/hip_runtime.h>
#include <hip/hip_bf16.h>

#define B_  32
#define T_  64
#define V_  30000
#define E_  64
#define H_  64
#define O_  7
#define BT_ 2048          // B_*T_
#define KS_ 8             // K-split ways for the embedding GEMM

typedef __attribute__((ext_vector_type(16))) __bf16 v16bf;
typedef __attribute__((ext_vector_type(8)))  float  v8f;
typedef __attribute__((ext_vector_type(4)))  float  v4f;
typedef __attribute__((ext_vector_type(2)))  float  v2f;

__device__ __forceinline__ float sigmoidf_(float x) { return 1.0f / (1.0f + __expf(-x)); }

// ---------------------------------------------------------------------------
// Kernel 1: embedding GEMM partials.  x(2048x30000) @ vocab(30000x64) with
// bf16 WMMA 16x16x32 (fp32 accum).  Wave = one (mtile, kchunk).  4 waves of a
// block share kchunk and stage the 32x64 vocab tile into LDS in fragment
// order.  Also produces per-row partial sums of x (for the 1/rowsum scaling).
// ---------------------------------------------------------------------------
template<bool TAIL>
__device__ __forceinline__ void k1_step(
    int k, int lane, const float* __restrict__ xrow,
    const float* __restrict__ vocab,
    __bf16 (*Bt)[32][16],           // [4][32][16] in LDS
    v8f* acc, float& rs)
{
  // Stage B tile: vocab rows k..k+31, cols 0..63 -> bf16, fragment layout:
  // Bt[c>>4][(c&15) + ((r&8)<<1)][(r&7) | ((r&16)>>1)]
  __syncthreads();
  for (int e = threadIdx.x; e < 32 * 64; e += 128) {
    int r = e >> 6, c = e & 63;
    float v = 0.0f;
    if (!TAIL || (k + r) < V_) v = vocab[(size_t)(k + r) * E_ + c];
    Bt[c >> 4][(c & 15) + ((r & 8) << 1)][(r & 7) | ((r & 16) >> 1)] = (__bf16)v;
  }
  __syncthreads();

  // A fragment per lane: row m = lane%16 (+mtile*16), K offsets
  // {base..base+7} U {base+16..base+23}, base = k + (lane>=16 ? 8 : 0).
  int base = k + ((lane & 16) ? 8 : 0);
  float af[16];
  if (!TAIL) {
    v4f a0 = *(const v4f*)(xrow + base);
    v4f a1 = *(const v4f*)(xrow + base + 4);
    v4f a2 = *(const v4f*)(xrow + base + 16);
    v4f a3 = *(const v4f*)(xrow + base + 20);
#pragma unroll
    for (int j = 0; j < 4; ++j) {
      af[j] = a0[j]; af[4 + j] = a1[j]; af[8 + j] = a2[j]; af[12 + j] = a3[j];
    }
  } else {
#pragma unroll
    for (int j = 0; j < 16; ++j) {
      int kk = base + (j < 8 ? j : j + 8);
      af[j] = (kk < V_) ? xrow[kk] : 0.0f;
    }
  }
  v16bf a;
#pragma unroll
  for (int j = 0; j < 16; ++j) { rs += af[j]; a[j] = (__bf16)af[j]; }

#pragma unroll
  for (int nt = 0; nt < 4; ++nt) {
    v16bf b = *(const v16bf*)&Bt[nt][lane][0];
    acc[nt] = __builtin_amdgcn_wmma_f32_16x16x32_bf16(
        false, a, false, b, (short)0, acc[nt], false, false);
  }
}

__global__ __launch_bounds__(128) void k1_embed(
    const float* __restrict__ x, const float* __restrict__ vocab,
    float* __restrict__ accP, float* __restrict__ rowsumP)
{
  __shared__ __align__(32) __bf16 Bt[4][32][16];
  int lane = threadIdx.x & 31;
  int wave = threadIdx.x >> 5;
  int kc   = blockIdx.x & 7;                 // shared by all 4 waves
  int mt   = (blockIdx.x >> 3) * 4 + wave;   // 0..127
  int m    = mt * 16 + (lane & 15);
  const float* xrow = x + (size_t)m * V_;

  v8f acc[4] = {};
  float rs = 0.0f;

  int k = kc * 32;
  for (; k + 32 <= V_; k += KS_ * 32)
    k1_step<false>(k, lane, xrow, vocab, Bt, acc, rs);
  if (k < V_)
    k1_step<true>(k, lane, xrow, vocab, Bt, acc, rs);

  // store the partial 16x64 tile (fully overwrites its slot -> no init pass)
  int n0 = lane & 15;
  int hi = (lane & 16) ? 8 : 0;
  float* ap = accP + ((size_t)kc * BT_ + mt * 16) * E_;
#pragma unroll
  for (int nt = 0; nt < 4; ++nt)
#pragma unroll
    for (int v = 0; v < 8; ++v)
      ap[(size_t)(v + hi) * E_ + nt * 16 + n0] = acc[nt][v];

  // rowsum: lane l and l+16 cover complementary K subsets of row m
  float ro = __shfl_xor(rs, 16);
  if (lane < 16) rowsumP[kc * BT_ + mt * 16 + lane] = rs + ro;
}

// ---------------------------------------------------------------------------
// Kernel 2a: embed = relu( (sum_c accP) / (sum_c rowsumP) )
// ---------------------------------------------------------------------------
__global__ __launch_bounds__(256) void k2a_final(
    const float* __restrict__ accP, const float* __restrict__ rowsumP,
    float* __restrict__ embed)
{
  int i = blockIdx.x * 256 + threadIdx.x;   // 0..BT_*E_-1
  int row = i >> 6;
  float a = 0.0f, s = 0.0f;
#pragma unroll
  for (int c = 0; c < KS_; ++c) {
    a += accP[(size_t)c * BT_ * E_ + i];
    s += rowsumP[c * BT_ + row];
  }
  float v = a / s;
  embed[i] = v > 0.0f ? v : 0.0f;
}

// ---------------------------------------------------------------------------
// Kernel 2b: G_g = embed @ W_i{r,z,n}^T + b_i{r,z,n}   (f32 WMMA 16x16x4)
// G layout: [3][BT_][H_]
// ---------------------------------------------------------------------------
__global__ __launch_bounds__(128) void k2b_gates(
    const float* __restrict__ embed,
    const float* __restrict__ Wir, const float* __restrict__ Wiz,
    const float* __restrict__ Win,
    const float* __restrict__ bir, const float* __restrict__ biz,
    const float* __restrict__ bin,
    float* __restrict__ G)
{
  int lane = threadIdx.x & 31;
  int wave = threadIdx.x >> 5;
  int mt   = blockIdx.x * 4 + wave;          // 0..127
  int n0   = lane & 15;
  int koff = (lane & 16) ? 2 : 0;
  int hi   = (lane & 16) ? 8 : 0;
  int row  = mt * 16 + n0;

  v2f a[16];
#pragma unroll
  for (int kc = 0; kc < 16; ++kc)
    a[kc] = *(const v2f*)(embed + (size_t)row * E_ + kc * 4 + koff);

  const float* W[3]  = {Wir, Wiz, Win};
  const float* bb[3] = {bir, biz, bin};
#pragma unroll
  for (int g = 0; g < 3; ++g) {
#pragma unroll
    for (int nt = 0; nt < 4; ++nt) {
      int n = nt * 16 + n0;
      v8f c = {};
#pragma unroll
      for (int kc = 0; kc < 16; ++kc) {
        v2f b = *(const v2f*)(W[g] + (size_t)n * E_ + kc * 4 + koff);  // B[k][n]=W[n][k]
        c = __builtin_amdgcn_wmma_f32_16x16x4_f32(
            false, a[kc], false, b, (short)0, c, false, false);
      }
      float bias = bb[g][n];
      float* Gp = G + ((size_t)g * BT_ + mt * 16) * E_;
#pragma unroll
      for (int v = 0; v < 8; ++v)
        Gp[(size_t)(v + hi) * E_ + n] = c[v] + bias;
    }
  }
}

// ---------------------------------------------------------------------------
// Kernel 3: persistent single-workgroup GRU scan + output projection.
// 8 waves: wave w owns tile (mt=w>>2, nt=w&3) of every 32x64 recurrent GEMM.
// Recurrent weights held in registers as B-fragments; h and gate results in LDS.
// ---------------------------------------------------------------------------
__global__ __launch_bounds__(256) void k3_gru(
    const float* __restrict__ G,
    const float* __restrict__ Whr, const float* __restrict__ Whz,
    const float* __restrict__ Whn,
    const float* __restrict__ bhr, const float* __restrict__ bhz,
    const float* __restrict__ bhn,
    const float* __restrict__ Wout, const float* __restrict__ bout,
    const int* __restrict__ ppred, float* __restrict__ out)
{
  __shared__ __align__(16) float hbuf[B_][H_];
  __shared__ __align__(16) float GH[3][B_][H_];   // h@Whr^T, h@Whz^T, h@Whn^T

  int tid  = threadIdx.x;
  int lane = tid & 31;
  int w    = tid >> 5;
  int mt   = w >> 2;
  int nt   = w & 3;
  int n0   = lane & 15;
  int koff = (lane & 16) ? 2 : 0;
  int hi   = (lane & 16) ? 8 : 0;
  int n    = nt * 16 + n0;
  int pp   = ppred[0];

  // recurrent weight B-fragments, loaded once: B[k][n] = W[n][k]
  v2f wb[3][16];
  const float* Wh[3] = {Whr, Whz, Whn};
#pragma unroll
  for (int g = 0; g < 3; ++g)
#pragma unroll
    for (int kc = 0; kc < 16; ++kc)
      wb[g][kc] = *(const v2f*)(Wh[g] + (size_t)n * H_ + kc * 4 + koff);

  // h0 = (1 - sigmoid(Gz[t=0])) * tanh(Gn[t=0])
  for (int i = tid; i < B_ * H_; i += 256) {
    int b = i >> 6, nn = i & 63;
    size_t r0 = (size_t)(b * T_) * H_ + nn;
    float z1 = sigmoidf_(G[(size_t)1 * BT_ * H_ + r0]);
    float n1 = tanhf(G[(size_t)2 * BT_ * H_ + r0]);
    hbuf[b][nn] = (1.0f - z1) * n1;
  }
  __syncthreads();

  for (int t = 1; t < pp; ++t) {
    // --- recurrent matmuls ---
    v2f a[16];
    int arow = mt * 16 + n0;
#pragma unroll
    for (int kc = 0; kc < 16; ++kc)
      a[kc] = *(const v2f*)&hbuf[arow][kc * 4 + koff];
#pragma unroll
    for (int g = 0; g < 3; ++g) {
      v8f c = {};
#pragma unroll
      for (int kc = 0; kc < 16; ++kc)
        c = __builtin_amdgcn_wmma_f32_16x16x4_f32(
            false, a[kc], false, wb[g][kc], (short)0, c, false, false);
#pragma unroll
      for (int v = 0; v < 8; ++v)
        GH[g][mt * 16 + v + hi][n] = c[v];
    }
    __syncthreads();

    // --- gate elementwise ---
    for (int i = tid; i < B_ * H_; i += 256) {
      int b = i >> 6, nn = i & 63;
      size_t rt = (size_t)(b * T_ + t) * H_ + nn;
      float hrv = GH[0][b][nn] + bhr[nn];                       // h@Whr^T + b_hr
      float r   = sigmoidf_(G[rt] + hrv);
      float z   = sigmoidf_(G[(size_t)1 * BT_ * H_ + rt] + GH[1][b][nn] + bhz[nn]);
      float nn2 = tanhf(G[(size_t)2 * BT_ * H_ + rt] + r * hrv);
      float h   = hbuf[b][nn];
      hbuf[b][nn] = (1.0f - z) * nn2 + z * h;
    }
    __syncthreads();
  }

  // output projection: (32x64) @ (64x7)^T + b_out
  for (int i = tid; i < B_ * O_; i += 256) {
    int b = i / O_, o = i % O_;
    float s = bout[o];
#pragma unroll
    for (int nn = 0; nn < H_; ++nn) s += hbuf[b][nn] * Wout[o * H_ + nn];
    out[i] = s;
  }
}

// ---------------------------------------------------------------------------
extern "C" void kernel_launch(void* const* d_in, const int* in_sizes, int n_in,
                              void* d_out, int out_size, void* d_ws, size_t ws_size,
                              hipStream_t stream)
{
  const float* x     = (const float*)d_in[0];
  const float* vocab = (const float*)d_in[1];
  const float* Wir   = (const float*)d_in[2];
  const float* Wiz   = (const float*)d_in[3];
  const float* Win   = (const float*)d_in[4];
  const float* Whr   = (const float*)d_in[5];
  const float* Whz   = (const float*)d_in[6];
  const float* Whn   = (const float*)d_in[7];
  const float* bir   = (const float*)d_in[8];
  const float* biz   = (const float*)d_in[9];
  const float* bin   = (const float*)d_in[10];
  const float* bhr   = (const float*)d_in[11];
  const float* bhz   = (const float*)d_in[12];
  const float* bhn   = (const float*)d_in[13];
  const float* Wout  = (const float*)d_in[14];
  const float* bout  = (const float*)d_in[15];
  const int*   pp    = (const int*)d_in[16];

  float* ws      = (float*)d_ws;
  float* accP    = ws;                              // KS_*BT_*E_
  float* rowsumP = accP + (size_t)KS_ * BT_ * E_;   // KS_*BT_
  float* embed   = rowsumP + KS_ * BT_;             // BT_*E_
  float* G       = embed + (size_t)BT_ * E_;        // 3*BT_*E_

  k1_embed <<<256, 128, 0, stream>>>(x, vocab, accP, rowsumP);
  k2a_final<<<(BT_ * E_) / 256, 256, 0, stream>>>(accP, rowsumP, embed);
  k2b_gates<<<32, 128, 0, stream>>>(embed, Wir, Wiz, Win, bir, biz, bin, G);
  k3_gru   <<<1, 256, 0, stream>>>(G, Whr, Whz, Whn, bhr, bhz, bhn,
                                   Wout, bout, pp, (float*)d_out);
}
